// GCRUCell_46686294507785
// MI455X (gfx1250) — matrix operand order repaired and checked by reading, not compiled, not run on hardware
//
#include <hip/hip_runtime.h>
#include <hip/hip_bf16.h>
#include <math.h>

typedef __attribute__((ext_vector_type(16))) __bf16 v16bf;
typedef __attribute__((ext_vector_type(8)))  __bf16 v8bf;
typedef __attribute__((ext_vector_type(8)))  float  v8f;

#define N_NODE  1024
#define N_BATCH 64
#define D_IN    8
#define H_DIM   64
#define C_DIM   72
#define NBC     (N_BATCH * C_DIM)     // 4608
#define K_PACK  384                   // 5*72 = 360 padded to mult of 32
#define N_ROWS  (N_BATCH * N_NODE)    // 65536

// ---------------------------------------------------------------------------
// CDNA5 async global->LDS copy (ASYNCcnt-tracked, no VGPR round trip).
// Each lane copies 16B from its global address to its LDS address.
// LDS operand: wave-relative LDS byte address (= low 32 bits of generic ptr).
// ---------------------------------------------------------------------------
__device__ __forceinline__ void async_load_b128(unsigned lds_off, const void* gaddr) {
  asm volatile("global_load_async_to_lds_b128 %0, %1, off"
               :
               : "v"(lds_off), "v"(gaddr)
               : "memory");
}

__device__ __forceinline__ void wait_async0() {
#if __has_builtin(__builtin_amdgcn_s_wait_asynccnt)
  __builtin_amdgcn_s_wait_asynccnt(0);
#else
  asm volatile("s_wait_asynccnt 0x0" ::: "memory");
#endif
}

// ---------------------------------------------------------------------------
// Elementwise / packing kernels (all feature panels stored K-major / "Bt")
// ---------------------------------------------------------------------------
__global__ void cast_f32_bf16_kernel(const float* __restrict__ src,
                                     __bf16* __restrict__ dst, int n) {
  int i = blockIdx.x * 256 + threadIdx.x;
  if (i < n) dst[i] = (__bf16)src[i];
}

// Packed projection weights, TRANSPOSED: Wt[j, r] (ncols x 384), bf16.
// r 0-71:    Wc0 + Wc3 - Wc2 - Wc5   (applied to X)
// r 72-143:  Wc1                     (Y1_g0)
// r 144-215: 2*Wc2                   (Y2_g0)
// r 216-287: Wc4                     (Y1_g1)
// r 288-359: 2*Wc5                   (Y2_g1)
// r 360-383: zero pad
__global__ void pack_proj_w(const float* __restrict__ W,
                            __bf16* __restrict__ Wt, int ncols) {
  int i = blockIdx.x * 256 + threadIdx.x;        // i = j*K_PACK + r
  int total = ncols * K_PACK;
  if (i >= total) return;
  int j = i / K_PACK, r = i % K_PACK;
  float v;
  if (r < 72) {
    v = W[(0 * 72 + r) * ncols + j] + W[(3 * 72 + r) * ncols + j]
      - W[(2 * 72 + r) * ncols + j] - W[(5 * 72 + r) * ncols + j];
  } else if (r < 144) {
    v = W[(1 * 72 + (r - 72)) * ncols + j];
  } else if (r < 216) {
    v = 2.0f * W[(2 * 72 + (r - 144)) * ncols + j];
  } else if (r < 288) {
    v = W[(4 * 72 + (r - 216)) * ncols + j];
  } else if (r < 360) {
    v = 2.0f * W[(5 * 72 + (r - 288)) * ncols + j];
  } else {
    v = 0.0f;
  }
  Wt[i] = (__bf16)v;
}

// Xpt[col, m] = concat(x, state)[b, m, c], col = b*72+c  (layout (NBC, N_NODE))
__global__ void pack_x_state(const float* __restrict__ x,
                             const float* __restrict__ state,
                             __bf16* __restrict__ Xpt) {
  int i = blockIdx.x * 256 + threadIdx.x;        // i = col*1024 + m
  if (i >= NBC * N_NODE) return;
  int col = i >> 10, m = i & 1023;
  int b = col / C_DIM, c = col % C_DIM;
  size_t row = (size_t)b * N_NODE + m;
  float v = (c < D_IN) ? x[row * D_IN + c]
                       : state[row * H_DIM + (c - D_IN)];
  Xpt[i] = (__bf16)v;
}

// Xpt[col, m] = concat(x, z*state)[b, m, c]; z = ZR[row, 0:64]
__global__ void pack_candidate(const float* __restrict__ x,
                               const float* __restrict__ state,
                               const float* __restrict__ ZR,
                               __bf16* __restrict__ Xpt) {
  int i = blockIdx.x * 256 + threadIdx.x;
  if (i >= NBC * N_NODE) return;
  int col = i >> 10, m = i & 1023;
  int b = col / C_DIM, c = col % C_DIM;
  size_t row = (size_t)b * N_NODE + m;
  float v;
  if (c < D_IN) {
    v = x[row * D_IN + c];
  } else {
    int h = c - D_IN;
    float z = ZR[row * 128 + h];               // z = z_r[..., :H]
    v = z * state[row * H_DIM + h];
  }
  Xpt[i] = (__bf16)v;
}

// F[(b*1024+n), s] = packed features; panels are transposed ([col][node])
__global__ void pack_feats(const __bf16* __restrict__ Xpt,
                           const __bf16* __restrict__ Y10t,
                           const __bf16* __restrict__ Y20t,
                           const __bf16* __restrict__ Y11t,
                           const __bf16* __restrict__ Y21t,
                           __bf16* __restrict__ F) {
  int i = blockIdx.x * 256 + threadIdx.x;
  if (i >= N_ROWS * K_PACK) return;
  int row = i / K_PACK, s = i % K_PACK;
  int b = row >> 10, n = row & 1023;
  __bf16 v;
  if      (s <  72) v = Xpt [((size_t)(b * C_DIM + s)        << 10) + n];
  else if (s < 144) v = Y10t[((size_t)(b * C_DIM + (s - 72)) << 10) + n];
  else if (s < 216) v = Y20t[((size_t)(b * C_DIM + (s - 144)) << 10) + n];
  else if (s < 288) v = Y11t[((size_t)(b * C_DIM + (s - 216)) << 10) + n];
  else if (s < 360) v = Y21t[((size_t)(b * C_DIM + (s - 288)) << 10) + n];
  else              v = (__bf16)0.0f;
  F[i] = v;
}

__global__ void act_sigmoid_bias(float* __restrict__ ZR,
                                 const float* __restrict__ gb) {
  int i = blockIdx.x * 256 + threadIdx.x;
  if (i >= N_ROWS * 128) return;
  float v = ZR[i] + gb[i & 127];
  ZR[i] = 1.0f / (1.0f + expf(-v));
}

__global__ void final_combine(const float* __restrict__ HC,
                              const float* __restrict__ ZR,
                              const float* __restrict__ state,
                              const float* __restrict__ ub,
                              float* __restrict__ out) {
  int i = blockIdx.x * 256 + threadIdx.x;
  if (i >= N_ROWS * H_DIM) return;
  int row = i >> 6, j = i & 63;
  float r  = ZR[(size_t)row * 128 + 64 + j];   // r = z_r[..., H:]
  float hc = tanhf(HC[i] + ub[j]);
  out[i] = r * state[i] + (1.0f - r) * hc;
}

// ---------------------------------------------------------------------------
// bf16 WMMA GEMM (NT):  C = A[M,K] @ Bt[N,K]^T      f32 accumulate
// Both operands K-major -> all tile staging is contiguous 64B rows moved with
// GLOBAL_LOAD_ASYNC_TO_LDS_B128 (double buffered, ASYNCcnt-synchronized).
// CT=true stores C transposed ([N][M], bf16) with one packed b128 per tile.
// Block = 128 threads = 4 waves.
// ---------------------------------------------------------------------------
template <int BM, int BN, int WROWS, int WCOLS, typename OutT, bool CT>
__global__ __launch_bounds__(128)
void gemm_nt(const __bf16* __restrict__ A, const __bf16* __restrict__ Bt,
             OutT* __restrict__ C, int M, int N, int K) {
  constexpr int WTM = BM / WROWS;   // rows per wave
  constexpr int WTN = BN / WCOLS;   // cols per wave
  constexpr int TM  = WTM / 16;
  constexpr int TN  = WTN / 16;

  __shared__ __bf16 As[2][BM][32] __attribute__((aligned(16)));
  __shared__ __bf16 Bs[2][BN][32] __attribute__((aligned(16)));  // [n][k]

  const int tid  = threadIdx.x;
  const int wave = tid >> 5;
  const int lane = tid & 31;
  const int wrow = wave / WCOLS;
  const int wcol = wave % WCOLS;
  const int rowBase = blockIdx.y * BM;
  const int colBase = blockIdx.x * BN;

  v8f acc[TM][TN];
#pragma unroll
  for (int mt = 0; mt < TM; ++mt)
#pragma unroll
    for (int nt = 0; nt < TN; ++nt)
#pragma unroll
      for (int r = 0; r < 8; ++r) acc[mt][nt][r] = 0.0f;

  // Stage one K-step (32) of A and B tiles via async b128 copies.
  // task t covers (row = t>>2, 16B chunk = t&3); 4 consecutive lanes move one
  // 64B row -> coalesced global, contiguous LDS.
  auto issueTiles = [&](int buf, int kb) {
    const int k0 = kb * 32;
    constexpr int AIT = (BM * 4) / 128;
#pragma unroll
    for (int it = 0; it < AIT; ++it) {
      int task = it * 128 + tid;
      int row = task >> 2, ch = task & 3;
      async_load_b128((unsigned)(size_t)&As[buf][row][ch * 8],
                      A + (size_t)(rowBase + row) * K + k0 + ch * 8);
    }
    constexpr int BIT = (BN * 4) / 128;
#pragma unroll
    for (int it = 0; it < BIT; ++it) {
      int task = it * 128 + tid;
      int row = task >> 2, ch = task & 3;
      async_load_b128((unsigned)(size_t)&Bs[buf][row][ch * 8],
                      Bt + (size_t)(colBase + row) * K + k0 + ch * 8);
    }
  };

  const int nk = K >> 5;
  issueTiles(0, 0);

  for (int kb = 0; kb < nk; ++kb) {
    wait_async0();        // loads issued last iteration have landed in LDS
    __syncthreads();      // visible to all waves
    if (kb + 1 < nk) issueTiles((kb + 1) & 1, kb + 1);
    const int buf = kb & 1;

    // A frag: lanes 0-15 -> K{0..7,16..23}, lanes 16-31 -> K{8..15,24..31}
    const int kA = ((lane >> 4) & 1) * 8;
    // B frag: lanes 0-15 -> K{0..7,16..23}, lanes 16-31 -> K{8..15,24..31}
    const int kB = ((lane >> 4) & 1) * 8;

    v16bf a[TM], b[TN];
#pragma unroll
    for (int mt = 0; mt < TM; ++mt) {
      const int row = wrow * WTM + mt * 16 + (lane & 15);
      v8bf lo = *reinterpret_cast<const v8bf*>(&As[buf][row][kA]);
      v8bf hi = *reinterpret_cast<const v8bf*>(&As[buf][row][kA + 16]);
      a[mt] = __builtin_shufflevector(lo, hi, 0, 1, 2, 3, 4, 5, 6, 7,
                                      8, 9, 10, 11, 12, 13, 14, 15);
    }
#pragma unroll
    for (int nt = 0; nt < TN; ++nt) {
      const int col = wcol * WTN + nt * 16 + (lane & 15);
      v8bf lo = *reinterpret_cast<const v8bf*>(&Bs[buf][col][kB]);
      v8bf hi = *reinterpret_cast<const v8bf*>(&Bs[buf][col][kB + 16]);
      b[nt] = __builtin_shufflevector(lo, hi, 0, 1, 2, 3, 4, 5, 6, 7,
                                      8, 9, 10, 11, 12, 13, 14, 15);
    }
#pragma unroll
    for (int mt = 0; mt < TM; ++mt)
#pragma unroll
      for (int nt = 0; nt < TN; ++nt)
        acc[mt][nt] = __builtin_amdgcn_wmma_f32_16x16x32_bf16(
            false, a[mt], false, b[nt], (short)0, acc[mt][nt], false, false);
  }

  // C/D layout: VGPR r -> M = r + (lane>=16 ? 8 : 0), N = lane & 15.
#pragma unroll
  for (int mt = 0; mt < TM; ++mt) {
#pragma unroll
    for (int nt = 0; nt < TN; ++nt) {
      const int col   = colBase + wcol * WTN + nt * 16 + (lane & 15);
      const int rbase = rowBase + wrow * WTM + mt * 16 + ((lane >> 4) << 3);
      if constexpr (CT) {
        // Transposed store: lane holds 8 consecutive rows of one column ->
        // one packed 16B store into Ct[col][rbase..rbase+7].
        union { __bf16 h[8]; uint4 q; } u;
#pragma unroll
        for (int r = 0; r < 8; ++r) u.h[r] = (__bf16)acc[mt][nt][r];
        *reinterpret_cast<uint4*>(
            reinterpret_cast<__bf16*>(C) + (size_t)col * M + rbase) = u.q;
      } else {
#pragma unroll
        for (int r = 0; r < 8; ++r)
          C[(size_t)(rbase + r) * N + col] = (OutT)acc[mt][nt][r];
      }
    }
  }
}

// ---------------------------------------------------------------------------
// Host-side orchestration
// ---------------------------------------------------------------------------
static inline int cdiv_host(int a, int b) { return (a + b - 1) / b; }

extern "C" void kernel_launch(void* const* d_in, const int* in_sizes, int n_in,
                              void* d_out, int out_size, void* d_ws, size_t ws_size,
                              hipStream_t stream) {
  (void)in_sizes; (void)n_in; (void)out_size; (void)ws_size;
  const float* x        = (const float*)d_in[0];
  const float* state    = (const float*)d_in[1];
  const float* graph0   = (const float*)d_in[2];
  const float* graph1   = (const float*)d_in[3];
  const float* gate_W   = (const float*)d_in[4];
  const float* gate_b   = (const float*)d_in[5];
  const float* update_W = (const float*)d_in[6];
  const float* update_b = (const float*)d_in[7];
  float* out = (float*)d_out;

  char* wsp = (char*)d_ws;
  size_t off = 0;
  auto carve = [&](size_t bytes) -> void* {
    void* p = (void*)(wsp + off);
    off += (bytes + 255) & ~(size_t)255;
    return p;
  };
  __bf16* Gb0  = (__bf16*)carve((size_t)N_NODE * N_NODE * 2);
  __bf16* Gb1  = (__bf16*)carve((size_t)N_NODE * N_NODE * 2);
  __bf16* Xpt  = (__bf16*)carve((size_t)NBC * N_NODE * 2);
  __bf16* Y10t = (__bf16*)carve((size_t)NBC * N_NODE * 2);
  __bf16* Y20t = (__bf16*)carve((size_t)NBC * N_NODE * 2);
  __bf16* Y11t = (__bf16*)carve((size_t)NBC * N_NODE * 2);
  __bf16* Y21t = (__bf16*)carve((size_t)NBC * N_NODE * 2);
  __bf16* F    = (__bf16*)carve((size_t)N_ROWS * K_PACK * 2);
  __bf16* Wgt  = (__bf16*)carve((size_t)128 * K_PACK * 2);
  __bf16* Wut  = (__bf16*)carve((size_t)64 * K_PACK * 2);
  float*  ZR   = (float*)carve((size_t)N_ROWS * 128 * 4);
  float*  HC   = (float*)carve((size_t)N_ROWS * 64 * 4);

  // --- prep: bf16 graphs + packed transposed weights ---
  {
    int n = N_NODE * N_NODE;
    cast_f32_bf16_kernel<<<cdiv_host(n, 256), 256, 0, stream>>>(graph0, Gb0, n);
    cast_f32_bf16_kernel<<<cdiv_host(n, 256), 256, 0, stream>>>(graph1, Gb1, n);
  }
  pack_proj_w<<<cdiv_host(128 * K_PACK, 256), 256, 0, stream>>>(gate_W,   Wgt, 128);
  pack_proj_w<<<cdiv_host(64 * K_PACK, 256), 256, 0, stream>>>(update_W, Wut, 64);

  dim3 gGraph(NBC / 128, N_NODE / 128);   // (36, 8)
  dim3 gProjG(128 / 128, N_ROWS / 128);   // (1, 512)
  dim3 gProjU(64 / 64,   N_ROWS / 128);   // (1, 512)

  // ---- gate path: z_r = sigmoid(agcn([x,state])) ----
  pack_x_state<<<cdiv_host(NBC * N_NODE, 256), 256, 0, stream>>>(x, state, Xpt);
  gemm_nt<128, 128, 2, 2, __bf16, true><<<gGraph, 128, 0, stream>>>(Gb0, Xpt,  Y10t, N_NODE, NBC, N_NODE);
  gemm_nt<128, 128, 2, 2, __bf16, true><<<gGraph, 128, 0, stream>>>(Gb0, Y10t, Y20t, N_NODE, NBC, N_NODE);
  gemm_nt<128, 128, 2, 2, __bf16, true><<<gGraph, 128, 0, stream>>>(Gb1, Xpt,  Y11t, N_NODE, NBC, N_NODE);
  gemm_nt<128, 128, 2, 2, __bf16, true><<<gGraph, 128, 0, stream>>>(Gb1, Y11t, Y21t, N_NODE, NBC, N_NODE);
  pack_feats<<<cdiv_host(N_ROWS * K_PACK, 256), 256, 0, stream>>>(Xpt, Y10t, Y20t, Y11t, Y21t, F);
  gemm_nt<128, 128, 2, 2, float, false><<<gProjG, 128, 0, stream>>>(F, Wgt, ZR, N_ROWS, 128, K_PACK);
  act_sigmoid_bias<<<cdiv_host(N_ROWS * 128, 256), 256, 0, stream>>>(ZR, gate_b);

  // ---- candidate path: hc = tanh(agcn([x, z*state])) ----
  pack_candidate<<<cdiv_host(NBC * N_NODE, 256), 256, 0, stream>>>(x, state, ZR, Xpt);
  gemm_nt<128, 128, 2, 2, __bf16, true><<<gGraph, 128, 0, stream>>>(Gb0, Xpt,  Y10t, N_NODE, NBC, N_NODE);
  gemm_nt<128, 128, 2, 2, __bf16, true><<<gGraph, 128, 0, stream>>>(Gb0, Y10t, Y20t, N_NODE, NBC, N_NODE);
  gemm_nt<128, 128, 2, 2, __bf16, true><<<gGraph, 128, 0, stream>>>(Gb1, Xpt,  Y11t, N_NODE, NBC, N_NODE);
  gemm_nt<128, 128, 2, 2, __bf16, true><<<gGraph, 128, 0, stream>>>(Gb1, Y11t, Y21t, N_NODE, NBC, N_NODE);
  pack_feats<<<cdiv_host(N_ROWS * K_PACK, 256), 256, 0, stream>>>(Xpt, Y10t, Y20t, Y11t, Y21t, F);
  gemm_nt<128, 64, 4, 1, float, false><<<gProjU, 128, 0, stream>>>(F, Wut, HC, N_ROWS, 64, K_PACK);

  // ---- h = r*state + (1-r)*hc ----
  final_combine<<<cdiv_host(N_ROWS * H_DIM, 256), 256, 0, stream>>>(HC, ZR, state, update_b, out);
}
